// Interaction_Attention_73126113182201
// MI455X (gfx1250) — matrix operand order repaired and checked by reading, not compiled, not run on hardware
//
#include <hip/hip_runtime.h>
#include <math.h>

#define NB   8192
#define SS   3
#define DD   2048
#define DKK  1024
#define GB   8          // batches per block
#define ROWS 48         // 6 * GB (3 img rows + 3 text rows per batch)
#define LDSK 2056       // 2048 + 8 bf16 pad -> row stride 4112 B, bank-conflict free

typedef __attribute__((ext_vector_type(16))) __bf16       v16bf;
typedef __attribute__((ext_vector_type(8)))  float        v8f;
typedef __attribute__((ext_vector_type(4)))  unsigned int u32x4;

union Frag { v16bf v; u32x4 q[2]; };

__device__ __forceinline__ unsigned short f2bf(float f) {
  unsigned int u = __float_as_uint(f);
  u = (u + 0x7FFFu + ((u >> 16) & 1u)) >> 16;   // round-to-nearest-even
  return (unsigned short)u;
}

// ---------------------------------------------------------------------------
// Kernel 0: convert [Wq;Wk] fp32 -> bf16 rows [2048][2048] in workspace (L2-hot)
// ---------------------------------------------------------------------------
__global__ void __launch_bounds__(256)
convert_w_kernel(const float* __restrict__ Wq, const float* __restrict__ Wk,
                 unsigned short* __restrict__ Wbf) {
  size_t base = ((size_t)blockIdx.x * 256 + threadIdx.x) * 4;   // 4 elems/thread
  const size_t half = (size_t)DKK * DD;                          // Wq | Wk boundary
  const float* src = (base < half) ? (Wq + base) : (Wk + (base - half));
  float4 v = *(const float4*)src;
  unsigned short* d = Wbf + base;
  d[0] = f2bf(v.x); d[1] = f2bf(v.y); d[2] = f2bf(v.z); d[3] = f2bf(v.w);
}

// ---------------------------------------------------------------------------
// Kernel 1: zero the score accumulators (ws is poisoned by harness)
// ---------------------------------------------------------------------------
__global__ void __launch_bounds__(256)
zero_scores_kernel(float* __restrict__ scores) {
  scores[(size_t)blockIdx.x * 256 + threadIdx.x] = 0.f;
}

// ---------------------------------------------------------------------------
// Kernel 2: fused projection (WMMA bf16) + 3x3 raw attention scores.
// Block = 256 threads (8 waves) = 8 batches. Never materializes Q/K in HBM.
// ---------------------------------------------------------------------------
__global__ void __launch_bounds__(256)
proj_score_kernel(const float* __restrict__ img, const float* __restrict__ text,
                  const unsigned short* __restrict__ Wbf,
                  const float* __restrict__ bq, const float* __restrict__ bk,
                  float* __restrict__ scores) {
  extern __shared__ char smem[];
  unsigned short* X  = (unsigned short*)smem;                       // [48][LDSK] bf16
  float*          QK = (float*)(smem + ROWS * LDSK * sizeof(unsigned short)); // [8][48][32]

  const int tid  = threadIdx.x;
  const int lane = tid & 31;
  const int wave = tid >> 5;
  const int h    = (lane >> 4) & 1;   // lane half selects K-segment
  const int ln   = lane & 15;
  const long batch0 = (long)blockIdx.x * GB;

  // ---- stage 48 rows of img/text into LDS as bf16 (read from HBM once) ----
  for (int e = tid; e < ROWS * (DD / 4); e += 256) {
    int r  = e >> 9;          // / 512 float4s per row
    int c4 = e & 511;
    int b  = r / 6, i = r % 6;
    const float* src = (i < 3)
        ? (img  + ((size_t)(batch0 + b) * SS + i)       * DD)
        : (text + ((size_t)(batch0 + b) * SS + (i - 3)) * DD);
    float4 v = ((const float4*)src)[c4];
    unsigned short* dst = X + r * LDSK + c4 * 4;
    dst[0] = f2bf(v.x); dst[1] = f2bf(v.y); dst[2] = f2bf(v.z); dst[3] = f2bf(v.w);
  }
  __syncthreads();

  float part[7];
#pragma unroll
  for (int i = 0; i < 7; ++i) part[i] = 0.f;

  float* myQK = QK + wave * (ROWS * 32);

  // ---- each wave covers 8 chunks of 16 DK-columns (8 waves * 8 * 16 = 1024) ----
  for (int c = 0; c < 8; ++c) {
    const int nq = (wave * 8 + c) * 16;
    v8f accQ0 = {}, accQ1 = {}, accQ2 = {};
    v8f accK0 = {}, accK1 = {}, accK2 = {};
    // B operand: B[k][n] = W[n][k]; lane ln owns output column nq+ln
    const unsigned short* wqrow = Wbf + (size_t)(nq + ln) * DD;
    const unsigned short* wkrow = Wbf + (size_t)(DKK + nq + ln) * DD;

    for (int k0 = 0; k0 < DD; k0 += 32) {
      const unsigned short* wq = wqrow + k0 + h * 8;    // segments [k0+8h, +8) and [k0+16+8h, +8)
      const unsigned short* wk = wkrow + k0 + h * 8;
      // WGP-scope prefetch (locality 3), two K-steps ahead so it completes
      // before the consuming clause; SYS-scope would stop at GL2 (useless here).
      __builtin_prefetch(wq + 64, 0, 3);
      __builtin_prefetch(wk + 64, 0, 3);
      Frag fbq, fbk;
      fbq.q[0] = *(const u32x4*)(wq);  fbq.q[1] = *(const u32x4*)(wq + 16);
      fbk.q[0] = *(const u32x4*)(wk);  fbk.q[1] = *(const u32x4*)(wk + 16);

      // A operand from LDS: segments [k0+8h, +8) and [k0+16+8h, +8)
      const unsigned short* x0 = X + (0 * 16 + ln) * LDSK + k0 + h * 8;
      const unsigned short* x1 = X + (1 * 16 + ln) * LDSK + k0 + h * 8;
      const unsigned short* x2 = X + (2 * 16 + ln) * LDSK + k0 + h * 8;
      Frag a0, a1, a2;
      a0.q[0] = *(const u32x4*)x0;  a0.q[1] = *(const u32x4*)(x0 + 16);
      a1.q[0] = *(const u32x4*)x1;  a1.q[1] = *(const u32x4*)(x1 + 16);
      a2.q[0] = *(const u32x4*)x2;  a2.q[1] = *(const u32x4*)(x2 + 16);

      accQ0 = __builtin_amdgcn_wmma_f32_16x16x32_bf16(false, a0.v, false, fbq.v, (short)0, accQ0, false, false);
      accK0 = __builtin_amdgcn_wmma_f32_16x16x32_bf16(false, a0.v, false, fbk.v, (short)0, accK0, false, false);
      accQ1 = __builtin_amdgcn_wmma_f32_16x16x32_bf16(false, a1.v, false, fbq.v, (short)0, accQ1, false, false);
      accK1 = __builtin_amdgcn_wmma_f32_16x16x32_bf16(false, a1.v, false, fbk.v, (short)0, accK1, false, false);
      accQ2 = __builtin_amdgcn_wmma_f32_16x16x32_bf16(false, a2.v, false, fbq.v, (short)0, accQ2, false, false);
      accK2 = __builtin_amdgcn_wmma_f32_16x16x32_bf16(false, a2.v, false, fbk.v, (short)0, accK2, false, false);
    }

    // ---- stage Q/K chunk (bias folded in) to this wave's LDS scratch ----
    const float biasq = bq[nq + ln];
    const float biask = bk[nq + ln];
#pragma unroll
    for (int r = 0; r < 8; ++r) {
      myQK[(0  + r + 8 * h) * 32 + ln]      = accQ0[r] + biasq;
      myQK[(0  + r + 8 * h) * 32 + 16 + ln] = accK0[r] + biask;
      myQK[(16 + r + 8 * h) * 32 + ln]      = accQ1[r] + biasq;
      myQK[(16 + r + 8 * h) * 32 + 16 + ln] = accK1[r] + biask;
      myQK[(32 + r + 8 * h) * 32 + ln]      = accQ2[r] + biasq;
      myQK[(32 + r + 8 * h) * 32 + 16 + ln] = accK2[r] + biask;
    }
    asm volatile("s_wait_dscnt 0" ::: "memory");   // wave-local cross-lane visibility

    // ---- accumulate 8 batches * 3 maps * 3x3 dot products over this chunk ----
    for (int it = lane; it < 216; it += 32) {
      int s   = it % 3;
      int q   = (it / 3) % 3;
      int map = (it / 9) % 3;
      int b   = it / 27;
      int rq = b * 6 + ((map == 2) ? 3 : 0) + q;  // it/ii: img q-rows, tt: text
      int rs = b * 6 + ((map == 1) ? 0 : 3) + s;  // ii: img s-rows, it/tt: text
      const float* Qp = myQK + rq * 32;
      const float* Kp = myQK + rs * 32 + 16;
      float sum = 0.f;
#pragma unroll
      for (int j = 0; j < 16; ++j) sum += Qp[j] * Kp[j];
      part[it >> 5] += sum;
    }
  }

  // ---- flush partials (waves race within a block -> atomic, L2-resident) ----
#pragma unroll
  for (int i = 0; i < 7; ++i) {
    int it = lane + i * 32;
    if (it < 216) {
      int s = it % 3, q = (it / 3) % 3, map = (it / 9) % 3, b = it / 27;
      atomicAdd(scores + ((((size_t)(batch0 + b)) * 3 + map) * 3 + q) * 3 + s, part[i]);
    }
  }
}

// ---------------------------------------------------------------------------
// Kernel 3: softmax*NORM, Linear(3,3), Bilinear(3,3,3) -> atten_i/atten_t [B][18]
// ---------------------------------------------------------------------------
__global__ void __launch_bounds__(256)
attn_small_kernel(const float* __restrict__ scores, const float* __restrict__ Wj,
                  const float* __restrict__ bj, const float* __restrict__ Wbil,
                  float* __restrict__ atten) {
  int b = blockIdx.x * 256 + threadIdx.x;
  if (b >= NB) return;
  const float* sc = scores + (size_t)b * 27;
  const float NORM = 0.03125f;  // 1/sqrt(1024)

  float sm[3][3][3];
#pragma unroll
  for (int m = 0; m < 3; ++m)
#pragma unroll
    for (int q = 0; q < 3; ++q) {
      float x0 = sc[(m * 3 + q) * 3 + 0];
      float x1 = sc[(m * 3 + q) * 3 + 1];
      float x2 = sc[(m * 3 + q) * 3 + 2];
      float mx = fmaxf(x0, fmaxf(x1, x2));
      float e0 = __expf(x0 - mx), e1 = __expf(x1 - mx), e2 = __expf(x2 - mx);
      float inv = NORM / (e0 + e1 + e2);
      sm[m][q][0] = e0 * inv; sm[m][q][1] = e1 * inv; sm[m][q][2] = e2 * inv;
    }

  float wj[9], bjv[3], wb[27];
#pragma unroll
  for (int i = 0; i < 9; ++i)  wj[i]  = Wj[i];
#pragma unroll
  for (int i = 0; i < 3; ++i)  bjv[i] = bj[i];
#pragma unroll
  for (int i = 0; i < 27; ++i) wb[i]  = Wbil[i];

  float jm[3][3][3];
#pragma unroll
  for (int m = 0; m < 3; ++m)
#pragma unroll
    for (int q = 0; q < 3; ++q)
#pragma unroll
      for (int o = 0; o < 3; ++o)
        jm[m][q][o] = wj[o * 3 + 0] * sm[m][q][0] + wj[o * 3 + 1] * sm[m][q][1] +
                      wj[o * 3 + 2] * sm[m][q][2] + bjv[o];

  float* out = atten + (size_t)b * 18;
#pragma unroll
  for (int q = 0; q < 3; ++q)
#pragma unroll
    for (int o = 0; o < 3; ++o) {
      float si = 0.f, st = 0.f;
#pragma unroll
      for (int i = 0; i < 3; ++i)
#pragma unroll
        for (int j = 0; j < 3; ++j) {
          float w = wb[o * 9 + i * 3 + j];
          si += jm[0][q][i] * w * jm[1][q][j];   // bilinear(j_it, j_ii)
          st += jm[0][q][i] * w * jm[2][q][j];   // bilinear(j_it, j_tt)
        }
      out[q * 3 + o]     = si;
      out[9 + q * 3 + o] = st;
    }
}

// ---------------------------------------------------------------------------
// Kernel 4: output_i = atten_i @ img, output_t = atten_t @ text (memory bound)
// ---------------------------------------------------------------------------
__global__ void __launch_bounds__(256)
out_kernel(const float* __restrict__ img, const float* __restrict__ text,
           const float* __restrict__ atten,
           float* __restrict__ out_i, float* __restrict__ out_t) {
  __shared__ float a[18];
  int b = blockIdx.x;
  if (threadIdx.x < 18) a[threadIdx.x] = atten[(size_t)b * 18 + threadIdx.x];
  __syncthreads();

  const float4* xi = (const float4*)(img  + (size_t)b * SS * DD);
  const float4* xt = (const float4*)(text + (size_t)b * SS * DD);
  float4* oi = (float4*)(out_i + (size_t)b * SS * DD);
  float4* ot = (float4*)(out_t + (size_t)b * SS * DD);
  const int R = DD / 4;   // 512 float4 per row

  for (int f = threadIdx.x; f < R; f += 256) {
    float4 r0 = xi[f], r1 = xi[R + f], r2 = xi[2 * R + f];
#pragma unroll
    for (int q = 0; q < 3; ++q) {
      float c0 = a[q * 3 + 0], c1 = a[q * 3 + 1], c2 = a[q * 3 + 2];
      float4 o;
      o.x = c0 * r0.x + c1 * r1.x + c2 * r2.x;
      o.y = c0 * r0.y + c1 * r1.y + c2 * r2.y;
      o.z = c0 * r0.z + c1 * r1.z + c2 * r2.z;
      o.w = c0 * r0.w + c1 * r1.w + c2 * r2.w;
      oi[q * R + f] = o;
    }
    float4 t0 = xt[f], t1 = xt[R + f], t2 = xt[2 * R + f];
#pragma unroll
    for (int q = 0; q < 3; ++q) {
      float c0 = a[9 + q * 3 + 0], c1 = a[9 + q * 3 + 1], c2 = a[9 + q * 3 + 2];
      float4 o;
      o.x = c0 * t0.x + c1 * t1.x + c2 * t2.x;
      o.y = c0 * t0.y + c1 * t1.y + c2 * t2.y;
      o.z = c0 * t0.z + c1 * t1.z + c2 * t2.z;
      o.w = c0 * t0.w + c1 * t1.w + c2 * t2.w;
      ot[q * R + f] = o;
    }
  }
}

// ---------------------------------------------------------------------------
extern "C" void kernel_launch(void* const* d_in, const int* in_sizes, int n_in,
                              void* d_out, int out_size, void* d_ws, size_t ws_size,
                              hipStream_t stream) {
  const float* img  = (const float*)d_in[0];
  const float* text = (const float*)d_in[1];
  const float* Wq   = (const float*)d_in[2];
  const float* bq   = (const float*)d_in[3];
  const float* Wk   = (const float*)d_in[4];
  const float* bk   = (const float*)d_in[5];
  const float* Wj   = (const float*)d_in[6];
  const float* bj   = (const float*)d_in[7];
  const float* Wbil = (const float*)d_in[8];

  float* out_i = (float*)d_out;
  float* out_t = out_i + (size_t)NB * SS * DD;

  // workspace layout: bf16 weights | scores | atten coefficients
  char* ws = (char*)d_ws;
  unsigned short* Wbf = (unsigned short*)ws;                       // 2048*2048*2 = 8 MB
  float* scores = (float*)(ws + (size_t)2 * DKK * DD * 2);         // NB*27*4
  float* atten  = scores + (size_t)NB * 27;                        // NB*18*4

  // (0) weight convert: 2048*2048 elems, 4/thread
  convert_w_kernel<<<(2 * DKK * DD) / (256 * 4), 256, 0, stream>>>(Wq, Wk, Wbf);
  // (1) zero scores: NB*27 = 221184 = 864 * 256
  zero_scores_kernel<<<(NB * 27) / 256, 256, 0, stream>>>(scores);
  // (2) fused WMMA projection + scores: 8 batches / block, 246528 B dynamic LDS
  size_t lds = (size_t)ROWS * LDSK * sizeof(unsigned short) + (size_t)8 * ROWS * 32 * sizeof(float);
  (void)hipFuncSetAttribute((const void*)proj_score_kernel,
                            hipFuncAttributeMaxDynamicSharedMemorySize, (int)lds);
  proj_score_kernel<<<NB / GB, 256, lds, stream>>>(img, text, Wbf, bq, bk, scores);
  // (3) tiny per-batch epilogue
  attn_small_kernel<<<NB / 256, 256, 0, stream>>>(scores, Wj, bj, Wbil, atten);
  // (4) memory-bound output recombination
  out_kernel<<<NB, 256, 0, stream>>>(img, text, atten, out_i, out_t);
}
